// SimpleSelfAttention_2052994367763
// MI455X (gfx1250) — matrix operand order repaired and checked by reading, not compiled, not run on hardware
//
#include <hip/hip_runtime.h>

typedef __attribute__((ext_vector_type(16))) __bf16 v16bf;
typedef __attribute__((ext_vector_type(8)))  __bf16 v8bf;
typedef __attribute__((ext_vector_type(4)))  __bf16 v4bf;
typedef __attribute__((ext_vector_type(8)))  float  v8f;
typedef __attribute__((ext_vector_type(4)))  float  f32x4;

#define S_LEN   4096
#define D_DIM   1024
#define MQ      32      // query rows per block
#define NK      64      // key rows per iteration
#define THREADS 256     // 8 wave32

// ---- dynamic LDS partition (bytes) ----
#define OFF_KT    0                         // [64][1024] bf16 = 131072
#define OFF_QT    131072                    // [1024][32] bf16 = 65536
#define OFF_ST    196608                    // [64][32] f32    = 8192
#define OFF_P     204800                    // [32][64] bf16   = 4096
#define OFF_PMAX  208896                    // [8][32] f32     = 1024
#define OFF_PSUM  209920                    // [8][32] f32     = 1024
#define OFF_M     210944                    // [32] f32
#define OFF_L     211072                    // [32] f32
#define OFF_SC    211200                    // [32] f32
#define SMEM_BYTES 211328

// A fragment: 16x32 bf16 tile, row-major source.
// lane m (0..15) holds row m; lanes 0-15 get K chunks [0..7],[16..23],
// lanes 16-31 get K chunks [8..15],[24..31].
__device__ __forceinline__ v16bf load_frag_a(const __bf16* base, int pitch,
                                             int row0, int k0, int lane) {
    const int hi = lane >> 4;
    const int m  = lane & 15;
    const __bf16* p = base + (row0 + m) * pitch + k0 + hi * 8;
    v8bf lo  = *(const v8bf*)(p);
    v8bf hiv = *(const v8bf*)(p + 16);
    return __builtin_shufflevector(lo, hiv, 0,1,2,3,4,5,6,7,8,9,10,11,12,13,14,15);
}

// B fragment: 32x16 bf16 tile [K x N], row-major source.
// lane l holds column n0 + l%16; lanes 0-15 get K [0..7],[16..23],
// lanes 16-31 get K [8..15],[24..31].
__device__ __forceinline__ v16bf load_frag_b(const __bf16* base, int pitch,
                                             int k0, int n0, int lane) {
    const int hi = lane >> 4;
    const __bf16* p = base + (k0 + hi * 8) * pitch + n0 + (lane & 15);
    v16bf r;
#pragma unroll
    for (int s = 0; s < 8; ++s) {
        r[s]     = p[s * pitch];
        r[s + 8] = p[(s + 16) * pitch];
    }
    return r;
}

__global__ void __launch_bounds__(THREADS, 1)
attn_flash_wmma(const float* __restrict__ x, float* __restrict__ out) {
    extern __shared__ char smem[];
    __bf16* Kt   = (__bf16*)(smem + OFF_KT);    // key/value tile, row-major [64][1024]
    __bf16* QT   = (__bf16*)(smem + OFF_QT);    // Q transposed [1024][32]
    float*  ST   = (float*) (smem + OFF_ST);    // S^T scores [64][32]
    __bf16* Pt   = (__bf16*)(smem + OFF_P);     // P row-major [32][64]
    float*  pmax = (float*) (smem + OFF_PMAX);
    float*  psum = (float*) (smem + OFF_PSUM);
    float*  mrow = (float*) (smem + OFF_M);
    float*  lrow = (float*) (smem + OFF_L);
    float*  srow = (float*) (smem + OFF_SC);

    const int t    = threadIdx.x;
    const int lane = t & 31;
    const int w    = t >> 5;          // wave id 0..7
    const int hi   = lane >> 4;
    const int b    = blockIdx.y;
    const int q0   = blockIdx.x * MQ;

    const float* xb = x + (size_t)b * S_LEN * D_DIM;

    // ---- load Q tile, convert fp32->bf16, store transposed QT[d][q] ----
    for (int i = 0; i < MQ; ++i) {
        f32x4 v = *(const f32x4*)(xb + (size_t)(q0 + i) * D_DIM + 4 * t);
#pragma unroll
        for (int e = 0; e < 4; ++e)
            QT[(4 * t + e) * MQ + i] = (__bf16)v[e];
    }
    if (t < MQ) { mrow[t] = -__builtin_inff(); lrow[t] = 0.0f; }

    const v8f zero = {0.f,0.f,0.f,0.f,0.f,0.f,0.f,0.f};
    v8f acc[2][8];
#pragma unroll
    for (int mt = 0; mt < 2; ++mt)
#pragma unroll
        for (int nt = 0; nt < 8; ++nt) acc[mt][nt] = zero;

    // ---- main loop over key blocks ----
    for (int kb = 0; kb < S_LEN / NK; ++kb) {
        const int k0 = kb * NK;
        __syncthreads();   // previous iteration finished reading Kt / Pt

        // load K/V tile (fp32 -> bf16), row-major into LDS; fully coalesced
        for (int i = 0; i < NK; ++i) {
            f32x4 v = *(const f32x4*)(xb + (size_t)(k0 + i) * D_DIM + 4 * t);
            v4bf c;
#pragma unroll
            for (int e = 0; e < 4; ++e) c[e] = (__bf16)v[e];
            *(v4bf*)(Kt + i * D_DIM + 4 * t) = c;
        }
        // prefetch next tile into L2 while we compute
        if (kb + 1 < S_LEN / NK) {
            const float* nx = xb + (size_t)(k0 + NK) * D_DIM;
#pragma unroll
            for (int r = 0; r < 8; ++r)
                __builtin_prefetch(nx + (size_t)t * 256 + r * 32, 0, 1);
        }
        __syncthreads();

        // ---- GEMM1: S^T[64][32] = Ktile @ Q^T ; wave -> one 16x16 subtile ----
        {
            const int kt = w & 3;       // key subtile
            const int qt = w >> 2;      // query subtile
            v8f c = zero;
#pragma unroll 4
            for (int d = 0; d < D_DIM; d += 32) {
                v16bf a  = load_frag_a(Kt, D_DIM, kt * 16, d, lane);
                v16bf bq = load_frag_b(QT, MQ, d, qt * 16, lane);
                c = __builtin_amdgcn_wmma_f32_16x16x32_bf16(
                        false, a, false, bq, (short)0, c, false, false);
            }
            const int col = qt * 16 + (lane & 15);
#pragma unroll
            for (int j = 0; j < 8; ++j)
                ST[(kt * 16 + hi * 8 + j) * MQ + col] = c[j];
        }
        __syncthreads();

        // ---- online softmax: pass 1 (partial max over 8-key groups) ----
        {
            const int q = t & 31, g = t >> 5;
            float pm = -__builtin_inff();
#pragma unroll
            for (int j = 0; j < 8; ++j)
                pm = fmaxf(pm, ST[(g * 8 + j) * MQ + q]);
            pmax[g * MQ + q] = pm;
        }
        __syncthreads();
        if (t < MQ) {
            float tm = pmax[t];
#pragma unroll
            for (int g = 1; g < 8; ++g) tm = fmaxf(tm, pmax[g * MQ + t]);
            const float mo = mrow[t];
            const float mn = fmaxf(mo, tm);
            srow[t] = __expf(mo - mn);   // rescale factor (0 on first block)
            mrow[t] = mn;
        }
        __syncthreads();

        // ---- pass 2: p = exp(s - m), write P bf16, partial sums ----
        {
            const int q = t & 31, g = t >> 5;
            const float mn = mrow[q];
            float ps = 0.0f;
#pragma unroll
            for (int j = 0; j < 8; ++j) {
                const float p = __expf(ST[(g * 8 + j) * MQ + q] - mn);
                ps += p;
                Pt[q * NK + g * 8 + j] = (__bf16)p;
            }
            psum[g * MQ + q] = ps;
        }
        __syncthreads();
        if (t < MQ) {
            float s = 0.0f;
#pragma unroll
            for (int g = 0; g < 8; ++g) s += psum[g * MQ + t];
            lrow[t] = lrow[t] * srow[t] + s;
        }
        __syncthreads();

        // ---- rescale accumulators by exp(m_old - m_new) per query row ----
#pragma unroll
        for (int mt = 0; mt < 2; ++mt) {
            float sc[8];
#pragma unroll
            for (int j = 0; j < 8; ++j) sc[j] = srow[mt * 16 + hi * 8 + j];
#pragma unroll
            for (int nt = 0; nt < 8; ++nt)
#pragma unroll
                for (int j = 0; j < 8; ++j) acc[mt][nt][j] *= sc[j];
        }

        // ---- GEMM2: O[32][128-slice] += P @ Vtile (Vtile == Kt, row-major) ----
#pragma unroll
        for (int kc = 0; kc < NK; kc += 32) {
            v16bf a0 = load_frag_a(Pt, NK, 0,  kc, lane);
            v16bf a1 = load_frag_a(Pt, NK, 16, kc, lane);
#pragma unroll
            for (int nt = 0; nt < 8; ++nt) {
                v16bf bv = load_frag_b(Kt, D_DIM, kc, w * 128 + nt * 16, lane);
                acc[0][nt] = __builtin_amdgcn_wmma_f32_16x16x32_bf16(
                                 false, a0, false, bv, (short)0, acc[0][nt], false, false);
                acc[1][nt] = __builtin_amdgcn_wmma_f32_16x16x32_bf16(
                                 false, a1, false, bv, (short)0, acc[1][nt], false, false);
            }
        }
    }

    __syncthreads();

    // ---- epilogue: divide by l, store fp32 output ----
    float* ob = out + (size_t)b * S_LEN * D_DIM;
#pragma unroll
    for (int mt = 0; mt < 2; ++mt) {
        float inv[8];
#pragma unroll
        for (int j = 0; j < 8; ++j)
            inv[j] = 1.0f / lrow[mt * 16 + hi * 8 + j];
#pragma unroll
        for (int nt = 0; nt < 8; ++nt) {
            const int col = w * 128 + nt * 16 + (lane & 15);
#pragma unroll
            for (int j = 0; j < 8; ++j) {
                const int row = q0 + mt * 16 + hi * 8 + j;
                ob[(size_t)row * D_DIM + col] = acc[mt][nt][j] * inv[j];
            }
        }
    }
}

extern "C" void kernel_launch(void* const* d_in, const int* in_sizes, int n_in,
                              void* d_out, int out_size, void* d_ws, size_t ws_size,
                              hipStream_t stream) {
    (void)in_sizes; (void)n_in; (void)d_ws; (void)ws_size; (void)out_size;
    const float* x = (const float*)d_in[0];
    float* out = (float*)d_out;

    // large CDNA5 LDS (320 KB/WGP): request ~206 KB dynamic shared
    hipFuncSetAttribute(reinterpret_cast<const void*>(attn_flash_wmma),
                        hipFuncAttributeMaxDynamicSharedMemorySize, SMEM_BYTES);

    dim3 grid(S_LEN / MQ, 4);
    attn_flash_wmma<<<grid, THREADS, SMEM_BYTES, stream>>>(x, out);
}
